// LSTM_75969381531775
// MI455X (gfx1250) — matrix operand order repaired and checked by reading, not compiled, not run on hardware
//
#include <hip/hip_runtime.h>
#include <hip/hip_bf16.h>

typedef __attribute__((ext_vector_type(16))) __bf16 v16bf;
typedef __attribute__((ext_vector_type(8)))  __bf16 v8bf;
typedef __attribute__((ext_vector_type(8)))  float  v8f;

#define T_STEPS 512
#define BATCH   64
#define INP     512
#define HID     1024
#define KTOT    1536
#define NGATE   4
#define NWG     64           // one WG per 16 h-columns
#define KC_TOT  48           // WMMA K-steps of 32 (48*32 = 1536)
#define CHUNK_K 64           // K per LDS chunk
#define CHUNKS  24           // 24 * 64 = 1536

__device__ __forceinline__ __bf16 f2bf(float f) {
    unsigned u = __builtin_bit_cast(unsigned, f);
    unsigned r = u + 0x7FFFu + ((u >> 16) & 1u);   // round-to-nearest-even
    unsigned short s = (unsigned short)(r >> 16);
    return __builtin_bit_cast(__bf16, s);
}

// branchless fast activations: single v_exp_f32 + v_rcp_f32, no div/saveexec
__device__ __forceinline__ float fast_rcp(float x)  { return __builtin_amdgcn_rcpf(x); }
__device__ __forceinline__ float fast_sig(float x)  { return fast_rcp(1.0f + __expf(-x)); }
__device__ __forceinline__ float fast_tanh(float x) {
    return 2.0f * fast_rcp(1.0f + __expf(-2.0f * x)) - 1.0f;
}

// ---------------------------------------------------------------------------
// Pack W[f|i|c|o] (each [1536,1024] f32, K-major) into a contiguous-k bf16
// B-operand lane layout: block = (gate, ntile, kc); per lane 16 contiguous
// bf16 = column n = nt*16+(lane&15), rows k = kc*32+(lane>>4)*16 .. +15.
// ---------------------------------------------------------------------------
__global__ __launch_bounds__(256) void pack_weights(
    const float* __restrict__ Wf, const float* __restrict__ Wi,
    const float* __restrict__ Wc, const float* __restrict__ Wo,
    __bf16* __restrict__ bp)
{
    int idx = blockIdx.x * 256 + threadIdx.x;        // one 16-elem lane chunk
    const int total = NGATE * NWG * KC_TOT * 32;
    if (idx >= total) return;
    int lane = idx & 31;
    int kc   = (idx >> 5) % KC_TOT;
    int nt   = ((idx >> 5) / KC_TOT) % NWG;
    int g    = (idx >> 5) / (KC_TOT * NWG);
    const float* W = (g == 0) ? Wf : (g == 1) ? Wi : (g == 2) ? Wc : Wo;
    int n  = nt * 16 + (lane & 15);
    int kb = kc * 32 + (lane >> 4) * 16;
    __bf16* dst = bp + (size_t)idx * 16;
#pragma unroll
    for (int i = 0; i < 16; ++i)
        dst[i] = f2bf(W[(size_t)(kb + i) * HID + n]);
}

__global__ __launch_bounds__(256) void conv_x(
    const float* __restrict__ x, __bf16* __restrict__ xbf, int n)
{
    int i = blockIdx.x * 256 + threadIdx.x;
    if (i < n) xbf[i] = f2bf(x[i]);
}

__global__ __launch_bounds__(256) void init_state(
    __bf16* __restrict__ hbf, float* __restrict__ cstate, unsigned* __restrict__ cnt)
{
    int i = blockIdx.x * 256 + threadIdx.x;
    if (i == 0) *cnt = 0u;
    if (i < 2 * BATCH * HID) hbf[i] = f2bf(0.0f);
    if (i < BATCH * HID)     cstate[i] = 0.0f;
}

// ---------------------------------------------------------------------------
// Persistent LSTM: 64 WGs x 256 threads (8 waves). WG j owns h-columns
// [16j,16j+16) of all 4 gates. wave = rt + 4*cpair; rt = batch row tile,
// cpair selects gate pair {f,i} / {c,o}. K=1536 in 24 double-buffered LDS
// chunks of 64, filled with global_load_async_to_lds_b128 (ASYNCcnt).
// One device-wide barrier per timestep.
// ---------------------------------------------------------------------------
__global__ __launch_bounds__(256) void lstm_persistent(
    const __bf16* __restrict__ xbf,     // [T][B][INP] bf16
    const __bf16* __restrict__ bpack,   // packed B operands
    __bf16* __restrict__ hbf,           // [2][B][H] bf16 ping-pong
    float*  __restrict__ cstate,        // [B][H] f32
    unsigned* __restrict__ cnt,         // grid barrier counter
    const float* __restrict__ bf_, const float* __restrict__ bi_,
    const float* __restrict__ bc_, const float* __restrict__ bo_,
    float* __restrict__ out)            // [T][B][H] + final h + final c
{
    __shared__ __align__(16) __bf16 abuf[2][BATCH][CHUNK_K];  // 2 x 8 KB A stage
    __shared__ __align__(16) __bf16 bbuf[2][8][512];          // 2 x 8 KB B stage
    __shared__ float gbuf[NGATE][BATCH][16];                  // 16 KB gates

    const int tid  = threadIdx.x;
    const int lane = tid & 31;
    const int wave = tid >> 5;
    const int rt   = wave & 3;            // row tile 0..3
    const int g0   = (wave >> 2) * 2;     // gates {g0, g0+1}
    const int wg   = blockIdx.x;          // h-column tile 0..63

    const int arow = rt * 16 + (lane & 15);     // A fragment row
    const int akb  = (lane >> 4) * 16;          // A fragment K sub-base
    const int ncol = lane & 15;                 // C/D fragment column
    const int mofs = (lane >> 4) * 8;           // C/D fragment row offset

    const float* bias0 = (g0 == 0) ? bf_ : bc_;
    const float* bias1 = (g0 == 0) ? bi_ : bo_;
    const float b0v = bias0[wg * 16 + ncol];
    const float b1v = bias1[wg * 16 + ncol];

    // gate g0 activation: sigmoid for f (g0==0), tanh for c~ (g0==2).
    // tanh(x) = 2*sigmoid(2x)-1 -> uniform scale/offset, no divergent branch.
    const float mA = (g0 == 2) ? 2.0f : 1.0f;
    const float sA = (g0 == 2) ? 2.0f : 1.0f;
    const float cA = (g0 == 2) ? -1.0f : 0.0f;

    // async stage of one K=64 chunk: A (64x64 bf16) + B (8 blocks of 512 bf16)
    auto issue_chunk = [&](int t, int ch) {
        int nb = ch & 1;
        const __bf16* base;
        int rstride;
        if (ch < 16) { base = hbf + (size_t)(t & 1) * BATCH * HID + ch * CHUNK_K; rstride = HID; }
        else         { base = xbf + (size_t)t * BATCH * INP + (ch * CHUNK_K - HID); rstride = INP; }
#pragma unroll
        for (int q = 0; q < 2; ++q) {
            int idx = tid + q * 256;              // 0..511 16B chunks
            int row = idx >> 3;
            int cc  = (idx & 7) * 8;
            unsigned           lds = (unsigned)(uintptr_t)&abuf[nb][row][cc];
            unsigned long long ga  = (unsigned long long)(base + row * rstride + cc);
            asm volatile("global_load_async_to_lds_b128 %0, %1, off"
                         :: "v"(lds), "v"(ga) : "memory");
        }
#pragma unroll
        for (int q = 0; q < 2; ++q) {
            int idx = tid + q * 256;              // 0..511 16B chunks
            int blk = idx >> 6;                   // 0..7 : g = blk>>1, kc-sub = blk&1
            int off = (idx & 63) * 8;
            int g   = blk >> 1;
            int kc  = ch * 2 + (blk & 1);
            const __bf16* src = bpack + ((size_t)(g * NWG + wg) * KC_TOT + kc) * 512 + off;
            unsigned           lds = (unsigned)(uintptr_t)&bbuf[nb][blk][off];
            unsigned long long ga  = (unsigned long long)src;
            asm volatile("global_load_async_to_lds_b128 %0, %1, off"
                         :: "v"(lds), "v"(ga) : "memory");
        }
    };

    for (int t = 0; t < T_STEPS; ++t) {
        // pull next step's x slice toward this WGP while we grind GEMMs
        __builtin_prefetch(xbf + (size_t)(t + 1) * BATCH * INP + tid * 128, 0, 1);

        issue_chunk(t, 0);
        v8f acc0 = {};
        v8f acc1 = {};
        for (int ch = 0; ch < CHUNKS; ++ch) {
            int nb = ch & 1;
            if (ch + 1 < CHUNKS) {
                issue_chunk(t, ch + 1);                         // 4 more async ops
                asm volatile("s_wait_asynccnt 4" ::: "memory"); // chunk ch landed
            } else {
                asm volatile("s_wait_asynccnt 0" ::: "memory");
            }
            __syncthreads();

#pragma unroll
            for (int kso = 0; kso < 2; ++kso) {
                union { v16bf v; v8bf h[2]; } a;
                const __bf16* ap = &abuf[nb][arow][kso * 32 + akb];
                a.h[0] = *(const v8bf*)(ap);         // K = kb .. kb+7
                a.h[1] = *(const v8bf*)(ap + 8);     // K = kb+8 .. kb+15
                v16bf b0 = *(const v16bf*)&bbuf[nb][(g0 + 0) * 2 + kso][lane * 16];
                v16bf b1 = *(const v16bf*)&bbuf[nb][(g0 + 1) * 2 + kso][lane * 16];
                acc0 = __builtin_amdgcn_wmma_f32_16x16x32_bf16(
                           false, a.v, false, b0, (short)0, acc0, false, false);
                acc1 = __builtin_amdgcn_wmma_f32_16x16x32_bf16(
                           false, a.v, false, b1, (short)0, acc1, false, false);
            }
            __syncthreads();  // all waves done with buf[nb] before it is refilled
        }

        // bias + activation -> LDS gate buffer (branchless, uniform per wave)
#pragma unroll
        for (int j = 0; j < 8; ++j) {
            int m = rt * 16 + mofs + j;
            float p0 = acc0[j] + b0v;
            float p1 = acc1[j] + b1v;
            float a0 = sA * fast_rcp(1.0f + __expf(-mA * p0)) + cA;
            float a1 = fast_sig(p1);             // gates i / o: sigmoid
            gbuf[g0 + 0][m][ncol] = a0;
            gbuf[g0 + 1][m][ncol] = a1;
        }
        __syncthreads();

        // fused LSTM cell update for this WG's 16 columns (exclusive ownership)
        __bf16* hdst = hbf + (size_t)((t + 1) & 1) * BATCH * HID;
#pragma unroll
        for (int q = 0; q < 4; ++q) {
            int li  = tid + q * 256;              // 0..1023
            int row = li >> 4;
            int n   = li & 15;
            int col = wg * 16 + n;
            float fg = gbuf[0][row][n];
            float ig = gbuf[1][row][n];
            float cg = gbuf[2][row][n];
            float og = gbuf[3][row][n];
            float cold = cstate[row * HID + col];
            float cnew = fg * cold + ig * cg;
            float hnew = og * fast_tanh(cnew);
            cstate[row * HID + col] = cnew;
            hdst[row * HID + col]   = f2bf(hnew);
            out[(size_t)t * BATCH * HID + row * HID + col] = hnew;
            if (t == T_STEPS - 1) {
                out[(size_t)T_STEPS * BATCH * HID + row * HID + col] = hnew;
                out[(size_t)T_STEPS * BATCH * HID + BATCH * HID + row * HID + col] = cnew;
            }
        }

        // device-wide barrier: release this step's h, wait for all 64 WGs
        __threadfence();
        __syncthreads();
        if (tid == 0) {
            __hip_atomic_fetch_add(cnt, 1u, __ATOMIC_RELEASE, __HIP_MEMORY_SCOPE_AGENT);
            unsigned target = (unsigned)gridDim.x * (unsigned)(t + 1);
            while (__hip_atomic_load(cnt, __ATOMIC_ACQUIRE, __HIP_MEMORY_SCOPE_AGENT) < target)
                __builtin_amdgcn_s_sleep(2);
        }
        __syncthreads();
        __threadfence();
    }
}

extern "C" void kernel_launch(void* const* d_in, const int* in_sizes, int n_in,
                              void* d_out, int out_size, void* d_ws, size_t ws_size,
                              hipStream_t stream)
{
    const float* x  = (const float*)d_in[0];
    const float* Wf = (const float*)d_in[1];
    const float* bf = (const float*)d_in[2];
    const float* Wi = (const float*)d_in[3];
    const float* bi = (const float*)d_in[4];
    const float* Wc = (const float*)d_in[5];
    const float* bc = (const float*)d_in[6];
    const float* Wo = (const float*)d_in[7];
    const float* bo = (const float*)d_in[8];
    float* out = (float*)d_out;

    char* ws = (char*)d_ws;
    size_t off = 0;
    __bf16* bpack = (__bf16*)(ws + off); off += (size_t)NGATE * NWG * KC_TOT * 512 * 2; // 12 MB
    __bf16* xbf   = (__bf16*)(ws + off); off += (size_t)T_STEPS * BATCH * INP * 2;      // 32 MB
    __bf16* hbf   = (__bf16*)(ws + off); off += (size_t)2 * BATCH * HID * 2;            // 256 KB
    float*  cst   = (float*) (ws + off); off += (size_t)BATCH * HID * 4;                // 256 KB
    unsigned* cnt = (unsigned*)(ws + off);

    const int packChunks = NGATE * NWG * KC_TOT * 32;   // 393216
    pack_weights<<<(packChunks + 255) / 256, 256, 0, stream>>>(Wf, Wi, Wc, Wo, bpack);

    const int xn = T_STEPS * BATCH * INP;               // 16.78M
    conv_x<<<(xn + 255) / 256, 256, 0, stream>>>(x, xbf, xn);

    init_state<<<(2 * BATCH * HID + 255) / 256, 256, 0, stream>>>(hbf, cst, cnt);

    lstm_persistent<<<NWG, 256, 0, stream>>>(xbf, bpack, hbf, cst, cnt,
                                             bf, bi, bc, bo, out);
}
